// MultiScalePeriodicAttention_62397284876456
// MI455X (gfx1250) — matrix operand order repaired and checked by reading, not compiled, not run on hardware
//
#include <hip/hip_runtime.h>
#include <hip/hip_bf16.h>
#include <math.h>

#define EMBED_DIM 256
#define SEQ_LEN   4096
#define BATCH     4

typedef _Float16 v8h  __attribute__((ext_vector_type(8)));
typedef _Float16 v16h __attribute__((ext_vector_type(16)));
typedef float    v8f  __attribute__((ext_vector_type(8)));

typedef unsigned int u32x4 __attribute__((ext_vector_type(4)));
typedef int          i32x4 __attribute__((ext_vector_type(4)));
typedef int          i32x8 __attribute__((ext_vector_type(8)));

#if defined(__has_builtin)
#if __has_builtin(__builtin_amdgcn_tensor_load_to_lds)
#define HAVE_TDM 1
#endif
#endif

union V16u { v16h v; v8h h[2]; };

__device__ __forceinline__ v16h load_pair(const _Float16* p0, const _Float16* p1) {
  V16u u;
  u.h[0] = *reinterpret_cast<const v8h*>(p0);
  u.h[1] = *reinterpret_cast<const v8h*>(p1);
  return u.v;
}

__device__ __forceinline__ v8f splat8(float x) {
  v8f r;
#pragma unroll
  for (int i = 0; i < 8; ++i) r[i] = x;
  return r;
}

// ---------------------------------------------------------------------------
// fp32 -> f16 conversion
// ---------------------------------------------------------------------------
__global__ void cvt_f32_f16(const float* __restrict__ in, _Float16* __restrict__ out, int n) {
  int i = blockIdx.x * blockDim.x + threadIdx.x;
  int stride = gridDim.x * blockDim.x;
  for (; i < n; i += stride) out[i] = (_Float16)in[i];
}

// ---------------------------------------------------------------------------
// QKV projection via v_wmma_f32_16x16x32_f16.
// Q,K row-major [b,s,d]; V transposed [b,d,s].
// Block = 128 thr (4 waves); wave computes 16 rows x 64 cols. grid=(S/16,3,B)
// ---------------------------------------------------------------------------
__global__ __launch_bounds__(128) void qkv_gemm(
    const _Float16* __restrict__ xh,
    const _Float16* __restrict__ Wqh, const _Float16* __restrict__ Wkh,
    const _Float16* __restrict__ Wvh,
    const float* __restrict__ bq, const float* __restrict__ bk,
    const float* __restrict__ bv,
    _Float16* __restrict__ Qh, _Float16* __restrict__ Kh,
    _Float16* __restrict__ Vth)
{
  const int D = EMBED_DIM, S = SEQ_LEN;
  const int lane = threadIdx.x & 31;
  const int wave = threadIdx.x >> 5;
  const int lr = lane & 15, hf = lane >> 4;
  const int s0 = blockIdx.x * 16;
  const int m  = blockIdx.y;
  const int b  = blockIdx.z;
  const _Float16* Wh = (m == 0) ? Wqh : (m == 1) ? Wkh : Wvh;
  const float* bias  = (m == 0) ? bq  : (m == 1) ? bk  : bv;
  const int e0 = wave * 64;

  v8f acc[4];
#pragma unroll
  for (int ct = 0; ct < 4; ++ct) acc[ct] = splat8(bias[e0 + ct * 16 + lr]);

  const _Float16* xb = xh + ((size_t)b * S + s0) * D;
#pragma unroll
  for (int dc = 0; dc < 8; ++dc) {
    const _Float16* ap = xb + (size_t)lr * D + dc * 32 + hf * 8;
    v16h a = load_pair(ap, ap + 16);
    v16h bmv[4];
#pragma unroll
    for (int ct = 0; ct < 4; ++ct) {
      const _Float16* bp = Wh + (size_t)(e0 + ct * 16 + lr) * D + dc * 32 + hf * 8;
      bmv[ct] = load_pair(bp, bp + 16);
    }
#pragma unroll
    for (int ct = 0; ct < 4; ++ct)
      acc[ct] = __builtin_amdgcn_wmma_f32_16x16x32_f16(false, a, false, bmv[ct],
                                                       (short)0, acc[ct], false, false);
  }

  if (m < 2) {
    _Float16* O = (m == 0) ? Qh : Kh;
#pragma unroll
    for (int ct = 0; ct < 4; ++ct)
#pragma unroll
      for (int v = 0; v < 8; ++v) {
        int row = v + hf * 8;
        O[((size_t)b * S + s0 + row) * D + e0 + ct * 16 + lr] = (_Float16)acc[ct][v];
      }
  } else {
#pragma unroll
    for (int ct = 0; ct < 4; ++ct) {
      v8h pk;
#pragma unroll
      for (int v = 0; v < 8; ++v) pk[v] = (_Float16)acc[ct][v];
      int e = e0 + ct * 16 + lr;
      *reinterpret_cast<v8h*>(Vth + ((size_t)b * D + e) * S + s0 + hf * 8) = pk;
    }
  }
}

// ---------------------------------------------------------------------------
// TDM 2D tile load: global (f16, row stride = stride0 elements) -> LDS,
// D# packed per CDNA5 ISA 8.3/8.4. One instruction per issuing wave
// (EXEC ignored by TDM). Tracked by TENSORcnt.
// amdgpu-toolchain (clang-23) 6-arg builtin form.
// ---------------------------------------------------------------------------
#ifdef HAVE_TDM
__device__ __forceinline__ void tdm_load_2d_f16(unsigned lds_addr, const void* gptr,
                                                unsigned tile_d0, unsigned tile_d1,
                                                unsigned long long stride0) {
  unsigned long long ga = (unsigned long long)(uintptr_t)gptr;
  u32x4 g0;
  g0[0] = 1u;                                              // count=1, load, no gather
  g0[1] = lds_addr;                                        // lds_addr [63:32]
  g0[2] = (unsigned)(ga & 0xFFFFFFFFu);                    // global_addr lo
  g0[3] = (unsigned)((ga >> 32) & 0x01FFFFFFu) | (2u << 30); // global_addr hi | type=2
  i32x8 g1;
  g1[0] = (int)(1u << 16);                                 // wg_mask=0, data_size=1 (2B)
  g1[1] = (int)((tile_d0 & 0xFFFFu) << 16);                // tensor_dim0 lo16
  g1[2] = (int)(((tile_d0 >> 16) & 0xFFFFu) |
                ((tile_d1 & 0xFFFFu) << 16));              // tensor_dim0 hi | tensor_dim1 lo
  g1[3] = (int)(((tile_d1 >> 16) & 0xFFFFu) |
                ((tile_d0 & 0xFFFFu) << 16));              // tensor_dim1 hi | tile_dim0
  g1[4] = (int)(tile_d1 & 0xFFFFu);                        // tile_dim1 | tile_dim2=0
  g1[5] = (int)(unsigned)(stride0 & 0xFFFFFFFFu);          // dim0_stride lo32
  g1[6] = (int)(unsigned)((stride0 >> 32) & 0xFFFFu);      // dim0_stride hi16 | dim1_stride lo=0
  g1[7] = 0;                                               // dim1_stride hi
  i32x4 z4 = {0, 0, 0, 0};
  i32x8 z8 = {0, 0, 0, 0, 0, 0, 0, 0};
  __builtin_amdgcn_tensor_load_to_lds(g0, g1, z4, z4, z8, 0);
}
#endif

// ---------------------------------------------------------------------------
// Flash attention with periodic bias.
// Block = 128 thr (4 waves); wave owns 16 query rows, full D=256 of O.
// K (64x256) and Vt (256x64) tiles double-buffered in dynamic LDS,
// filled by TDM async tensor loads (wave 0 issues, s_wait_tensorcnt sync).
// grid = (S/64, B), dynamic LDS = 128 KB.
// ---------------------------------------------------------------------------
__global__ __launch_bounds__(128) void flash_attn(
    const _Float16* __restrict__ Qh, const _Float16* __restrict__ Kh,
    const _Float16* __restrict__ Vth, const float* __restrict__ beta,
    float* __restrict__ out)
{
  const int D = EMBED_DIM, S = SEQ_LEN;
  const int NT = S / 64;

  extern __shared__ _Float16 smem[];
  _Float16* KbL = smem;                    // [2][64][256]
  _Float16* VbL = smem + 2 * 64 * 256;     // [2][256][64]

  __shared__ float bias_tab[720];
  __shared__ __align__(16) _Float16 Pbuf[4][16][64];

  {
    float b0 = beta[0], b1 = beta[1];
    const float TWO_PI = 6.28318530717958647692f;
    for (int t = threadIdx.x; t < 720; t += blockDim.x) {
      float ft = (float)t;
      bias_tab[t] = b0 * __cosf(TWO_PI * ft / 24.0f)
                  + b1 * __cosf(TWO_PI * ft / 720.0f);
    }
  }

  const int lane = threadIdx.x & 31;
  const int wave = threadIdx.x >> 5;
  const int lr = lane & 15, hf = lane >> 4;
  const int b  = blockIdx.y;
  const int i0 = blockIdx.x * 64 + wave * 16;

  const _Float16* Kg = Kh + (size_t)b * S * D;   // [S][256]
  const _Float16* Vg = Vth + (size_t)b * D * S;  // [256][S]

#ifdef HAVE_TDM
  const unsigned kbase = (unsigned)(uintptr_t)(void*)KbL;
  const unsigned vbase = (unsigned)(uintptr_t)(void*)VbL;
  if (wave == 0) {
    // prologue: tiles 0 and 1 in flight
    tdm_load_2d_f16(kbase,                 Kg,             256, 64, 256);
    tdm_load_2d_f16(vbase,                 Vg,              64, 256, S);
    tdm_load_2d_f16(kbase + 64 * 256 * 2,  Kg + 64 * 256,  256, 64, 256);
    tdm_load_2d_f16(vbase + 256 * 64 * 2,  Vg + 64,         64, 256, S);
  }
#endif

  // Q tile (16x256) in registers, A-layout chunks of K=32
  v16h qa[8];
  {
    const _Float16* qb = Qh + ((size_t)b * S + i0 + lr) * D;
#pragma unroll
    for (int dc = 0; dc < 8; ++dc) {
      const _Float16* p = qb + dc * 32 + hf * 8;
      qa[dc] = load_pair(p, p + 16);
    }
  }

  v8f o[16];
#pragma unroll
  for (int ct = 0; ct < 16; ++ct) o[ct] = splat8(0.0f);
  float rowm[8], rowl[8];
#pragma unroll
  for (int v = 0; v < 8; ++v) { rowm[v] = -INFINITY; rowl[v] = 0.0f; }

  const float scale = 0.0625f; // 1/sqrt(256)

  for (int it = 0; it < NT; ++it) {
    const int j0 = it * 64;
    const int cur = it & 1;
    _Float16* Kt = KbL + cur * 64 * 256;   // [64][256]
    _Float16* Vt = VbL + cur * 256 * 64;   // [256][64]

#ifdef HAVE_TDM
    if (wave == 0) {
      if (it + 1 < NT) __builtin_amdgcn_s_wait_tensorcnt(2); // tile 'it' retired
      else             __builtin_amdgcn_s_wait_tensorcnt(0);
    }
    __syncthreads();
#else
    __syncthreads();
    // cooperative fallback: K tile 64x256, V tile 256x64
    for (int r = threadIdx.x >> 1; r < 64; r += 64) {
      int c = (threadIdx.x & 1) * 128;
#pragma unroll
      for (int q = 0; q < 16; ++q)
        *reinterpret_cast<v8h*>(Kt + r * 256 + c + q * 8) =
            *reinterpret_cast<const v8h*>(Kg + (size_t)(j0 + r) * 256 + c + q * 8);
    }
    for (int r = threadIdx.x >> 3; r < 256; r += 16) {
      int c = (threadIdx.x & 7) * 8;
      *reinterpret_cast<v8h*>(Vt + r * 64 + c) =
          *reinterpret_cast<const v8h*>(Vg + (size_t)r * S + j0 + c);
    }
    __syncthreads();
#endif

    // ---- logits S = Q K^T from LDS K tile ----
    v8f s[4];
#pragma unroll
    for (int jt = 0; jt < 4; ++jt) {
      v8f sa = splat8(0.0f);
      const _Float16* kcol = Kt + (jt * 16 + lr) * 256 + hf * 8;
#pragma unroll
      for (int dc = 0; dc < 8; ++dc) {
        v16h kb = load_pair(kcol + dc * 32, kcol + dc * 32 + 16);
        sa = __builtin_amdgcn_wmma_f32_16x16x32_f16(false, qa[dc], false, kb,
                                                    (short)0, sa, false, false);
      }
      s[jt] = sa;
    }

    // ---- scale + periodic bias (cos table, period 720) ----
#pragma unroll
    for (int jt = 0; jt < 4; ++jt) {
      int base = i0 + hf * 8 - (j0 + jt * 16 + lr) + 4320; // >= 0
      int bm = base % 720;
#pragma unroll
      for (int v = 0; v < 8; ++v) {
        int idx = bm + v;
        if (idx >= 720) idx -= 720;
        s[jt][v] = s[jt][v] * scale + bias_tab[idx];
      }
    }

    // ---- online softmax ----
    float tm[8];
#pragma unroll
    for (int v = 0; v < 8; ++v)
      tm[v] = fmaxf(fmaxf(s[0][v], s[1][v]), fmaxf(s[2][v], s[3][v]));
#pragma unroll
    for (int off = 1; off < 16; off <<= 1)
#pragma unroll
      for (int v = 0; v < 8; ++v)
        tm[v] = fmaxf(tm[v], __shfl_xor(tm[v], off, 32));

    float alpha[8];
#pragma unroll
    for (int v = 0; v < 8; ++v) {
      float mn = fmaxf(rowm[v], tm[v]);
      alpha[v] = __expf(rowm[v] - mn);
      rowm[v] = mn;
    }

    float rs[8];
#pragma unroll
    for (int v = 0; v < 8; ++v) rs[v] = 0.0f;
#pragma unroll
    for (int jt = 0; jt < 4; ++jt)
#pragma unroll
      for (int v = 0; v < 8; ++v) {
        float p = __expf(s[jt][v] - rowm[v]);
        s[jt][v] = p;
        rs[v] += p;
      }
#pragma unroll
    for (int off = 1; off < 16; off <<= 1)
#pragma unroll
      for (int v = 0; v < 8; ++v)
        rs[v] += __shfl_xor(rs[v], off, 32);
#pragma unroll
    for (int v = 0; v < 8; ++v) rowl[v] = rowl[v] * alpha[v] + rs[v];

#pragma unroll
    for (int ct = 0; ct < 16; ++ct)
#pragma unroll
      for (int v = 0; v < 8; ++v) o[ct][v] *= alpha[v];

    // ---- P (C/D layout) -> LDS row-major f16 ----
#pragma unroll
    for (int jt = 0; jt < 4; ++jt)
#pragma unroll
      for (int v = 0; v < 8; ++v)
        Pbuf[wave][v + hf * 8][jt * 16 + lr] = (_Float16)s[jt][v];
    __syncthreads();

    // ---- O += P * V from LDS Vt tile ----
#pragma unroll
    for (int jc = 0; jc < 2; ++jc) {
      const _Float16* pp = &Pbuf[wave][lr][jc * 32 + hf * 8];
      v16h pa = load_pair(pp, pp + 16);
      const _Float16* vcol = Vt + jc * 32 + hf * 8;
#pragma unroll
      for (int ct = 0; ct < 16; ++ct) {
        const _Float16* vp = vcol + (ct * 16 + lr) * 64;
        v16h vb = load_pair(vp, vp + 16);
        o[ct] = __builtin_amdgcn_wmma_f32_16x16x32_f16(false, pa, false, vb,
                                                       (short)0, o[ct], false, false);
      }
    }
    __syncthreads();   // everyone done with buf[cur] before it is re-filled

#ifdef HAVE_TDM
    if (wave == 0 && it + 2 < NT) {
      const int j2 = (it + 2) * 64;
      tdm_load_2d_f16(kbase + cur * 64 * 256 * 2, Kg + (size_t)j2 * 256, 256, 64, 256);
      tdm_load_2d_f16(vbase + cur * 256 * 64 * 2, Vg + j2,                64, 256, S);
    }
#endif
  }

  // ---- epilogue ----
  float* ob = out + ((size_t)b * S + i0) * D;
#pragma unroll
  for (int v = 0; v < 8; ++v) {
    float inv = 1.0f / rowl[v];
    int row = v + hf * 8;
#pragma unroll
    for (int ct = 0; ct < 16; ++ct)
      ob[(size_t)row * D + ct * 16 + lr] = o[ct][v] * inv;
  }
}

// ---------------------------------------------------------------------------
// Launch
// ---------------------------------------------------------------------------
extern "C" void kernel_launch(void* const* d_in, const int* in_sizes, int n_in,
                              void* d_out, int out_size, void* d_ws, size_t ws_size,
                              hipStream_t stream) {
  const float* x    = (const float*)d_in[0];
  const float* Wq   = (const float*)d_in[1];
  const float* bq   = (const float*)d_in[2];
  const float* Wk   = (const float*)d_in[3];
  const float* bk   = (const float*)d_in[4];
  const float* Wv   = (const float*)d_in[5];
  const float* bv   = (const float*)d_in[6];
  const float* beta = (const float*)d_in[7];
  float* out = (float*)d_out;

  const size_t S = SEQ_LEN, D = EMBED_DIM, B = BATCH;
  const size_t nx = B * S * D;
  const size_t nw = D * D;

  char* ws = (char*)d_ws;
  size_t off = 0;
  _Float16* xh  = (_Float16*)(ws + off); off += nx * 2;
  _Float16* Wqh = (_Float16*)(ws + off); off += nw * 2;
  _Float16* Wkh = (_Float16*)(ws + off); off += nw * 2;
  _Float16* Wvh = (_Float16*)(ws + off); off += nw * 2;
  _Float16* Qh  = (_Float16*)(ws + off); off += nx * 2;
  _Float16* Kh  = (_Float16*)(ws + off); off += nx * 2;
  _Float16* Vth = (_Float16*)(ws + off); off += nx * 2;  // [b, d, s]

  cvt_f32_f16<<<2048, 256, 0, stream>>>(x, xh, (int)nx);
  cvt_f32_f16<<<256, 256, 0, stream>>>(Wq, Wqh, (int)nw);
  cvt_f32_f16<<<256, 256, 0, stream>>>(Wk, Wkh, (int)nw);
  cvt_f32_f16<<<256, 256, 0, stream>>>(Wv, Wvh, (int)nw);

  qkv_gemm<<<dim3(S / 16, 3, B), 128, 0, stream>>>(
      xh, Wqh, Wkh, Wvh, bq, bk, bv, Qh, Kh, Vth);

  const size_t dyn_lds = (2 * 64 * 256 + 2 * 256 * 64) * sizeof(_Float16); // 128 KB
  flash_attn<<<dim3(S / 64, B), 128, dyn_lds, stream>>>(Qh, Kh, Vth, beta, out);
}
